// ConvolutionLayer_37297495999125
// MI455X (gfx1250) — matrix operand order repaired and checked by reading, not compiled, not run on hardware
//
#include <hip/hip_runtime.h>
#include <hip/hip_bf16.h>

typedef __attribute__((ext_vector_type(16))) __bf16 v16bf;
typedef __attribute__((ext_vector_type(8)))  float  v8f;

#define BSZ   8
#define CIN   64
#define TLEN  32768
#define RCH   64
#define CCOND 80
#define CPAD  96      // cond channels padded to multiple of 32
#define TT    128     // time tile per workgroup
#define HALO  16      // (K-1)*DIL causal left context

#define WMMA_BF16(A, Bv, C) \
  __builtin_amdgcn_wmma_f32_16x16x32_bf16(false, (A), false, (Bv), (short)0, (C), false, false)

__device__ __forceinline__ float fast_sigmoid(float x) { return 1.f / (1.f + __expf(-x)); }
__device__ __forceinline__ float fast_tanh(float x) {
  float e = __expf(2.f * x);
  return (e - 1.f) / (e + 1.f);
}

// A-fragment element -> K index mapping for 16-bit A (16x32), contiguous order matching B:
// element e, lane-half hf:  K = 16*hf + e
__device__ __forceinline__ int a_k_of(int e, int hf) {
  return 16 * hf + e;
}

__global__ __launch_bounds__(256, 1) void wavenet_layer_kernel(
    const float* __restrict__ x, const float* __restrict__ cond,
    const float* __restrict__ wconv, const float* __restrict__ bconv,
    const float* __restrict__ wout, const float* __restrict__ bout,
    const float* __restrict__ wcond,
    float* __restrict__ outMain, float* __restrict__ outSkip)
{
  // time-major bf16 tiles: B-fragment load = one contiguous 32B per lane
  __shared__ __align__(32) __bf16 xs[TT + HALO][CIN];  // 18.0 KB
  __shared__ __align__(32) __bf16 cs[TT][CPAD];        // 24.0 KB
  __shared__ __align__(32) __bf16 zs[TT][RCH];         // 16.0 KB

  const int tid  = threadIdx.x;
  const int lane = tid & 31;
  const int wv   = tid >> 5;
  const int nIdx = lane & 15;   // N (time) within 16x16 tile / A row M
  const int hf   = lane >> 4;   // lane half
  const int b    = blockIdx.y;
  const int t0   = blockIdx.x * TT;

  // ---------------- stage activations -> LDS (fp32 -> bf16, transpose to [t][c]) --------
  for (int i = tid; i < (TT + HALO) * CIN; i += 256) {
    int c  = i / (TT + HALO);
    int tt = i - c * (TT + HALO);
    int gt = t0 - HALO + tt;
    float v = (gt >= 0) ? x[((size_t)b * CIN + c) * TLEN + gt] : 0.f;  // causal zero pad
    xs[tt][c] = (__bf16)v;
  }
  for (int i = tid; i < CCOND * TT; i += 256) {
    int c  = i / TT;
    int tt = i - c * TT;
    cs[tt][c] = (__bf16)cond[((size_t)b * CCOND + c) * TLEN + t0 + tt];
  }
  for (int i = tid; i < (CPAD - CCOND) * TT; i += 256) {
    int c  = CCOND + i / TT;
    int tt = i % TT;
    cs[tt][c] = (__bf16)0.f;
  }
  // warm next tile (gfx1250 global_prefetch_b8; speculative, OOB-safe)
  if (tid < CIN && t0 + TT < TLEN)
    __builtin_prefetch(&x[((size_t)b * CIN + tid) * TLEN + t0 + TT], 0, 1);

  // ---------------- build weight A-fragments in registers (L2-resident weights) --------
  const int p    = wv & 3;   // channel-pair block: filter rows 16p.., gate rows 64+16p..
  const int nset = wv >> 2;  // which half of the 8 time blocks this wave owns

  v16bf aConv[2][3][2];      // [filter|gate][tap][kstep]
  v16bf aCond[2][3];
  #pragma unroll
  for (int side = 0; side < 2; ++side) {
    const int row = side * RCH + 16 * p + nIdx;   // output channel (A-matrix row M)
    #pragma unroll
    for (int tap = 0; tap < 3; ++tap)
      #pragma unroll
      for (int ks = 0; ks < 2; ++ks) {
        v16bf a;
        #pragma unroll
        for (int e = 0; e < 16; ++e) {
          int k = a_k_of(e, hf);
          a[e] = (__bf16)wconv[((size_t)row * CIN + (32 * ks + k)) * 3 + tap];
        }
        aConv[side][tap][ks] = a;
      }
    #pragma unroll
    for (int ks = 0; ks < 3; ++ks) {
      v16bf a;
      #pragma unroll
      for (int e = 0; e < 16; ++e) {
        int cc = 32 * ks + a_k_of(e, hf);
        a[e] = (__bf16)((cc < CCOND) ? wcond[(size_t)row * CCOND + cc] : 0.f);
      }
      aCond[side][ks] = a;
    }
  }
  float biasF[8], biasG[8];
  #pragma unroll
  for (int v = 0; v < 8; ++v) {
    int ch = 16 * p + 8 * hf + v;
    biasF[v] = bconv[ch];
    biasG[v] = bconv[RCH + ch];
  }

  __syncthreads();

  // ---------------- stage 1: y = Wconv(*)x + Wcond*cond + b ; gated activation ---------
  #pragma unroll
  for (int nbi = 0; nbi < 4; ++nbi) {
    const int nb = nset * 4 + nbi;
    v8f accF = {};
    v8f accG = {};
    #pragma unroll
    for (int tap = 0; tap < 3; ++tap) {        // dilated taps: input row shift +8*tap
      #pragma unroll
      for (int ks = 0; ks < 2; ++ks) {
        v16bf bx = *(const v16bf*)&xs[nb * 16 + nIdx + 8 * tap][32 * ks + 16 * hf];
        accF = WMMA_BF16(aConv[0][tap][ks], bx, accF);
        accG = WMMA_BF16(aConv[1][tap][ks], bx, accG);
      }
    }
    #pragma unroll
    for (int ks = 0; ks < 3; ++ks) {
      v16bf bc = *(const v16bf*)&cs[nb * 16 + nIdx][32 * ks + 16 * hf];
      accF = WMMA_BF16(aCond[0][ks], bc, accF);
      accG = WMMA_BF16(aCond[1][ks], bc, accG);
    }
    const int t = t0 + nb * 16 + nIdx;
    #pragma unroll
    for (int v = 0; v < 8; ++v) {
      int chF = 16 * p + 8 * hf + v;           // C/D layout: row = 8*hf + v
      float z = fast_tanh(accF[v] + biasF[v]) * fast_sigmoid(accG[v] + biasG[v]);
      outSkip[((size_t)b * RCH + chF) * TLEN + t] = z;   // skip output (fp32)
      zs[nb * 16 + nIdx][chF] = (__bf16)z;               // restage for stage-2 GEMM
    }
  }

  __syncthreads();

  // ---------------- stage 2: output = Wout @ z + bias_out ------------------------------
  const int m2 = wv & 3;                       // 64 rows -> 4 M blocks
  v16bf aOut[2];
  #pragma unroll
  for (int ks = 0; ks < 2; ++ks) {
    v16bf a;
    #pragma unroll
    for (int e = 0; e < 16; ++e) {
      int k = a_k_of(e, hf);
      a[e] = (__bf16)wout[(size_t)(16 * m2 + nIdx) * RCH + 32 * ks + k];
    }
    aOut[ks] = a;
  }
  float biasO[8];
  #pragma unroll
  for (int v = 0; v < 8; ++v) biasO[v] = bout[16 * m2 + 8 * hf + v];

  #pragma unroll
  for (int nbi = 0; nbi < 4; ++nbi) {
    const int nb = nset * 4 + nbi;
    v8f acc = {};
    #pragma unroll
    for (int ks = 0; ks < 2; ++ks) {
      v16bf bz = *(const v16bf*)&zs[nb * 16 + nIdx][32 * ks + 16 * hf];
      acc = WMMA_BF16(aOut[ks], bz, acc);
    }
    const int t = t0 + nb * 16 + nIdx;
    #pragma unroll
    for (int v = 0; v < 8; ++v) {
      int ch = 16 * m2 + 8 * hf + v;
      outMain[((size_t)b * RCH + ch) * TLEN + t] = acc[v] + biasO[v];
    }
  }
}

extern "C" void kernel_launch(void* const* d_in, const int* in_sizes, int n_in,
                              void* d_out, int out_size, void* d_ws, size_t ws_size,
                              hipStream_t stream) {
  const float* x     = (const float*)d_in[0];
  const float* cond  = (const float*)d_in[1];
  const float* wconv = (const float*)d_in[2];
  const float* bconv = (const float*)d_in[3];
  const float* wout  = (const float*)d_in[4];
  const float* bout  = (const float*)d_in[5];
  const float* wcond = (const float*)d_in[6];

  float* outMain = (float*)d_out;                            // (B, R, T)
  float* outSkip = outMain + (size_t)BSZ * RCH * TLEN;       // (B, R, T)

  dim3 grid(TLEN / TT, BSZ);   // 256 x 8 = 2048 workgroups
  wavenet_layer_kernel<<<grid, 256, 0, stream>>>(x, cond, wconv, bconv,
                                                 wout, bout, wcond,
                                                 outMain, outSkip);
}
